// AdapterBank_47639777247802
// MI455X (gfx1250) — matrix operand order repaired and checked by reading, not compiled, not run on hardware
//
#include <hip/hip_runtime.h>

// ---------------- problem constants ----------------
constexpr int B_  = 8;
constexpr int N_  = 1024;
constexpr int TD_ = 5120;
constexpr int SD_ = 2048;
constexpr int BD_ = 1024;
constexpr int K_  = 16;     // queries
constexpr int M_  = 4;      // specialists
constexpr int H_  = 8;
constexpr int HD_ = 256;
constexpr int R_  = B_ * N_;   // 8192 token rows

// ---------------- types ----------------
typedef __attribute__((ext_vector_type(16))) __bf16        v16bf;
typedef __attribute__((ext_vector_type(8)))  float         v8f;
typedef __attribute__((ext_vector_type(4)))  unsigned int  u32x4;

union Frag { v16bf v; u32x4 q[2]; };

__device__ __forceinline__ float bf2f(unsigned short u) {
    unsigned int x = ((unsigned int)u) << 16;
    return __builtin_bit_cast(float, x);
}
__device__ __forceinline__ unsigned short f2bf(float f) {
    unsigned int x = __builtin_bit_cast(unsigned int, f);
    x += 0x7fffu + ((x >> 16) & 1u);          // round-to-nearest-even
    return (unsigned short)(x >> 16);
}
__device__ __forceinline__ float gelu_exact(float v) {
    return 0.5f * v * (1.0f + erff(v * 0.70710678118654752f));
}

// CDNA5 async LDS DMA: 16B global -> LDS, tracked by ASYNCcnt (no VGPR bounce)
__device__ __forceinline__ void async_copy16(unsigned int ldsOff, const void* gptr) {
    asm volatile("global_load_async_to_lds_b128 %0, %1, off"
                 :: "v"(ldsOff), "v"(gptr) : "memory");
}
__device__ __forceinline__ void wait_async0() {
    asm volatile("s_wait_asynccnt 0x0" ::: "memory");
}

// ---------------- bf16 convert (straight) ----------------
__global__ void cvt_bf16(const float* __restrict__ src, unsigned short* __restrict__ dst, int n) {
    int i = blockIdx.x * 256 + threadIdx.x;
    if (i < n) dst[i] = f2bf(src[i]);
}

// ---------------- bf16 convert + transpose: dst[n*rows + k] = bf16(src[k*cols + n]) ----------------
__global__ __launch_bounds__(256) void cvt_bf16_t(
    const float* __restrict__ src, unsigned short* __restrict__ dst, int rows, int cols) {
    __shared__ float tile[32][33];
    int bx = blockIdx.x * 32;              // n block
    int by = blockIdx.y * 32;              // k block
    int tx = threadIdx.x & 31, ty = threadIdx.x >> 5;   // 32 x 8
#pragma unroll
    for (int i = 0; i < 32; i += 8)
        tile[ty + i][tx] = src[(size_t)(by + ty + i) * cols + bx + tx];
    __syncthreads();
#pragma unroll
    for (int i = 0; i < 32; i += 8)
        dst[(size_t)(bx + ty + i) * rows + by + tx] = f2bf(tile[tx][ty + i]);
}

// ---------------- WMMA GEMM: C = A(bf16,MxK) * BT(bf16,NxK)^T + bias ----------------
// block tile 128x128, K-step 32, 8 waves; wave tile 64x32 (4x2 accums)
// double-buffered LDS fed by global_load_async_to_lds_b128
#define BM 128
#define BN 128
#define BK 32
#define LDP 40   // LDS row pitch (bf16 elems); 80B stride, keeps 16B alignment

template<int GELU_OUT, int OUT_BF16>
__global__ __launch_bounds__(256) void wmma_gemm(
    const unsigned short* __restrict__ A, const unsigned short* __restrict__ BT,
    const float* __restrict__ bias, float* __restrict__ Cf,
    unsigned short* __restrict__ Cb, int Ntot, int Ktot)
{
    __shared__ __attribute__((aligned(16))) unsigned short sA[2][BM * LDP];  // [m][k]
    __shared__ __attribute__((aligned(16))) unsigned short sB[2][BN * LDP];  // [n][k]

    const int tid   = threadIdx.x;
    const int n0    = blockIdx.x * BN;
    const int m0    = blockIdx.y * BM;
    const int wid   = tid >> 5, lane = tid & 31;
    const int waveM = wid >> 2, waveN = wid & 3;
    const int mW    = waveM * 64, nW = waveN * 32;
    const int half  = lane >> 4, lr = lane & 15;

    // staging: each thread owns 16 contiguous elements (2 x b128) of A tile and of B tile
    const int srow = tid >> 1;                 // 0..127
    const int scol = (tid & 1) * 16;           // 0 or 16
    const unsigned short* gA = A  + (size_t)(m0 + srow) * Ktot + scol;
    const unsigned short* gB = BT + (size_t)(n0 + srow) * Ktot + scol;
    const unsigned int sOff = (unsigned int)(srow * LDP + scol) * 2u;
    const unsigned int aBase[2] = { (unsigned int)(uintptr_t)&sA[0][0],
                                    (unsigned int)(uintptr_t)&sA[1][0] };
    const unsigned int bBase[2] = { (unsigned int)(uintptr_t)&sB[0][0],
                                    (unsigned int)(uintptr_t)&sB[1][0] };

    v8f acc[4][2];
    const v8f vzero = {0.f,0.f,0.f,0.f,0.f,0.f,0.f,0.f};
#pragma unroll
    for (int i = 0; i < 4; ++i)
#pragma unroll
        for (int j = 0; j < 2; ++j) acc[i][j] = vzero;

    // prologue: DMA tile 0 into buffer 0
    async_copy16(aBase[0] + sOff,       gA);
    async_copy16(aBase[0] + sOff + 16u, gA + 8);
    async_copy16(bBase[0] + sOff,       gB);
    async_copy16(bBase[0] + sOff + 16u, gB + 8);
    wait_async0();
    __syncthreads();

    int buf = 0;
    for (int kt = 0; kt < Ktot; kt += BK) {
        // DMA next tile into the other buffer while we compute on this one
        if (kt + BK < Ktot) {
            const int nb = buf ^ 1;
            const unsigned short* ga = gA + kt + BK;
            const unsigned short* gb = gB + kt + BK;
            async_copy16(aBase[nb] + sOff,       ga);
            async_copy16(aBase[nb] + sOff + 16u, ga + 8);
            async_copy16(bBase[nb] + sOff,       gb);
            async_copy16(bBase[nb] + sOff + 16u, gb + 8);
        }

        // fragments per ISA 16-bit layouts
        const unsigned short* cA = &sA[buf][0];
        const unsigned short* cB = &sB[buf][0];
        Frag af[4], bf[2];
#pragma unroll
        for (int mt = 0; mt < 4; ++mt) {
            int row = mW + mt * 16 + lr;
            af[mt].q[0] = *(const u32x4*)(cA + row * LDP + half * 8);       // K {0..7}/{8..15}
            af[mt].q[1] = *(const u32x4*)(cA + row * LDP + 16 + half * 8);  // K {16..23}/{24..31}
        }
#pragma unroll
        for (int nt = 0; nt < 2; ++nt) {
            int col = nW + nt * 16 + lr;
            bf[nt].q[0] = *(const u32x4*)(cB + col * LDP + half * 8);       // K {0..7}/{8..15}
            bf[nt].q[1] = *(const u32x4*)(cB + col * LDP + 16 + half * 8);  // K {16..23}/{24..31}
        }
#pragma unroll
        for (int mt = 0; mt < 4; ++mt)
#pragma unroll
            for (int nt = 0; nt < 2; ++nt)
                acc[mt][nt] = __builtin_amdgcn_wmma_f32_16x16x32_bf16(
                    false, af[mt].v, false, bf[nt].v, (short)0, acc[mt][nt], false, false);

        wait_async0();      // own DMA done ...
        __syncthreads();    // ... and everyone's, before buffers swap
        buf ^= 1;
    }

    // ---- epilogue: bias (+GELU) and store fp32 or bf16 ----
#pragma unroll
    for (int mt = 0; mt < 4; ++mt) {
#pragma unroll
        for (int nt = 0; nt < 2; ++nt) {
            int   gc = n0 + nW + nt * 16 + lr;
            float bv = bias[gc];
#pragma unroll
            for (int e = 0; e < 8; ++e) {
                int   gr = m0 + mW + mt * 16 + half * 8 + e;
                float v  = acc[mt][nt][e] + bv;
                if (GELU_OUT) v = gelu_exact(v);
                if (OUT_BF16) Cb[(size_t)gr * Ntot + gc] = f2bf(v);
                else          Cf[(size_t)gr * Ntot + gc] = v;
            }
        }
    }
}

// ---------------- LayerNorm (fp32 in -> bf16 out), row = SD_ ----------------
__global__ __launch_bounds__(256) void ln_bf16(
    const float* __restrict__ X, const float* __restrict__ g,
    const float* __restrict__ bt, unsigned short* __restrict__ Y)
{
    __shared__ float r1[256], r2[256];
    int row = blockIdx.x, tid = threadIdx.x;
    const float* x = X + (size_t)row * SD_;
    float vals[8], s = 0.f, s2 = 0.f;
#pragma unroll
    for (int i = 0; i < 8; ++i) {
        float v = x[tid + i * 256]; vals[i] = v; s += v; s2 += v * v;
    }
    r1[tid] = s; r2[tid] = s2; __syncthreads();
    for (int off = 128; off; off >>= 1) {
        if (tid < off) { r1[tid] += r1[tid + off]; r2[tid] += r2[tid + off]; }
        __syncthreads();
    }
    float mean = r1[0] * (1.f / SD_);
    float var  = r2[0] * (1.f / SD_) - mean * mean;
    float inv  = rsqrtf(var + 1e-5f);
#pragma unroll
    for (int i = 0; i < 8; ++i) {
        int c = tid + i * 256;
        Y[(size_t)row * SD_ + c] = f2bf((vals[i] - mean) * inv * g[c] + bt[c]);
    }
}

// ---------------- residual + LayerNorm (fp32 out, remapped rows) ----------------
__global__ __launch_bounds__(256) void ln_res(
    const float* __restrict__ X, const float* __restrict__ qm,
    const float* __restrict__ g, const float* __restrict__ bt,
    float* __restrict__ dst, int outerStride)
{
    __shared__ float r1[256], r2[256];
    int row = blockIdx.x, tid = threadIdx.x;   // row = b*K_ + k
    int b = row / K_, k = row % K_;
    const float* x  = X  + (size_t)row * SD_;
    const float* qr = qm + (size_t)k   * SD_;
    float vals[8], s = 0.f, s2 = 0.f;
#pragma unroll
    for (int i = 0; i < 8; ++i) {
        int c = tid + i * 256;
        float v = x[c] + qr[c]; vals[i] = v; s += v; s2 += v * v;
    }
    r1[tid] = s; r2[tid] = s2; __syncthreads();
    for (int off = 128; off; off >>= 1) {
        if (tid < off) { r1[tid] += r1[tid + off]; r2[tid] += r2[tid + off]; }
        __syncthreads();
    }
    float mean = r1[0] * (1.f / SD_);
    float var  = r2[0] * (1.f / SD_) - mean * mean;
    float inv  = rsqrtf(var + 1e-5f);
    float* d = dst + (size_t)b * outerStride + (size_t)k * SD_;
#pragma unroll
    for (int i = 0; i < 8; ++i) {
        int c = tid + i * 256;
        d[c] = (vals[i] - mean) * inv * g[c] + bt[c];
    }
}

// ---------------- q projection: QP(K_ x SD_) = q @ wq + bq (fp32) ----------------
__global__ __launch_bounds__(256) void qproj(
    const float* __restrict__ q, const float* __restrict__ wq,
    const float* __restrict__ bq, float* __restrict__ QP)
{
    int o = blockIdx.x * 256 + threadIdx.x;    // < K_*SD_
    int k = o / SD_, j = o % SD_;
    float s = bq[j];
    for (int i = 0; i < SD_; ++i) s += q[k * SD_ + i] * wq[(size_t)i * SD_ + j];
    QP[o] = s;
}

// ---------------- attention per (b,h): scores->softmax->ctx ----------------
__global__ __launch_bounds__(256) void attn(
    const float* __restrict__ QP, const unsigned short* __restrict__ KP,
    const unsigned short* __restrict__ VP, float* __restrict__ CTX)
{
    __shared__ float sc[K_ * N_];              // 64 KB (LDS is 320KB/WGP on CDNA5)
    int b = blockIdx.x >> 3, h = blockIdx.x & 7;
    int tid = threadIdx.x;

    // scores[k][n] = (qp . kp) / sqrt(HD)
    for (int o = tid; o < K_ * N_; o += 256) {
        int k = o >> 10, n = o & (N_ - 1);
        const float*          qr = QP + (size_t)k * SD_ + h * HD_;
        const unsigned short* kr = KP + ((size_t)(b * N_ + n)) * SD_ + h * HD_;
        float s = 0.f;
        for (int d = 0; d < HD_; d += 8) {
            unsigned short t[8];
            *(u32x4*)t = *(const u32x4*)(kr + d);
#pragma unroll
            for (int u = 0; u < 8; ++u) s += qr[d + u] * bf2f(t[u]);
        }
        sc[o] = s * 0.0625f;                   // 1/sqrt(256)
    }
    __syncthreads();

    // softmax over n: 16 lanes per row, shuffle reduce (wave32 -> 2 rows/wave)
    {
        int row = tid >> 4, j = tid & 15;
        float mx = -3.4e38f;
        for (int n = j; n < N_; n += 16) mx = fmaxf(mx, sc[row * N_ + n]);
        for (int off = 8; off; off >>= 1) mx = fmaxf(mx, __shfl_xor(mx, off, 16));
        float sum = 0.f;
        for (int n = j; n < N_; n += 16) {
            float e = __expf(sc[row * N_ + n] - mx);
            sc[row * N_ + n] = e; sum += e;
        }
        for (int off = 8; off; off >>= 1) sum += __shfl_xor(sum, off, 16);
        float inv = 1.f / sum;
        for (int n = j; n < N_; n += 16) sc[row * N_ + n] *= inv;
    }
    __syncthreads();

    // ctx[k][d] = att[k][:] . vp[:, d]
    for (int o = tid; o < K_ * HD_; o += 256) {
        int k = o >> 8, d = o & (HD_ - 1);
        const unsigned short* vc = VP + (size_t)(b * N_) * SD_ + h * HD_ + d;
        const float* ar = &sc[k * N_];
        float s = 0.f;
        for (int n = 0; n < N_; ++n) s += ar[n] * bf2f(vc[(size_t)n * SD_]);
        CTX[((size_t)b * K_ + k) * SD_ + h * HD_ + d] = s;
    }
}

// ---------------- router: mean pool ----------------
__global__ void pool_mean(const float* __restrict__ h, float* __restrict__ pooled) {
    int idx = blockIdx.x * 256 + threadIdx.x;  // < B_*TD_
    int b = idx / TD_, t = idx % TD_;
    const float* p = h + (size_t)b * N_ * TD_ + t;
    float s = 0.f;
    for (int n = 0; n < N_; ++n) s += p[(size_t)n * TD_];
    pooled[idx] = s * (1.f / N_);
}

// ---------------- router MLP + softmax ----------------
__global__ __launch_bounds__(512) void router(
    const float* __restrict__ pooled, const float* __restrict__ w1,
    const float* __restrict__ b1, const float* __restrict__ w2,
    const float* __restrict__ b2, float* __restrict__ probs, float* __restrict__ outp)
{
    __shared__ float hid[512];
    __shared__ float lgts[M_];
    int b = blockIdx.x, tid = threadIdx.x;
    const float* pr = pooled + (size_t)b * TD_;
    float s = b1[tid];
    for (int i = 0; i < TD_; ++i) s += pr[i] * w1[(size_t)i * 512 + tid];
    hid[tid] = gelu_exact(s);
    __syncthreads();
    if (tid < M_) {
        float t = b2[tid];
        for (int j = 0; j < 512; ++j) t += hid[j] * w2[j * M_ + tid];
        lgts[tid] = t;
    }
    __syncthreads();
    if (tid == 0) {
        float mx = lgts[0];
        for (int m = 1; m < M_; ++m) mx = fmaxf(mx, lgts[m]);
        float e[M_], sum = 0.f;
        for (int m = 0; m < M_; ++m) { e[m] = __expf(lgts[m] - mx); sum += e[m]; }
        for (int m = 0; m < M_; ++m) {
            float p = e[m] / sum;
            probs[b * M_ + m] = p;
            outp[b * M_ + m]  = p;
        }
    }
}

// ---------------- top-2 mixture combine into c_agg second half ----------------
__global__ void combine(const float* __restrict__ probs, const float* __restrict__ cspec,
                        float* __restrict__ out)
{
    int o = blockIdx.x * 256 + threadIdx.x;    // < B_*K_*SD_
    int b   = o / (K_ * SD_);
    int rem = o % (K_ * SD_);
    float p[M_];
#pragma unroll
    for (int m = 0; m < M_; ++m) p[m] = probs[b * M_ + m];
    int i0 = 0;
    for (int m = 1; m < M_; ++m) if (p[m] > p[i0]) i0 = m;
    int i1 = -1;
    for (int m = 0; m < M_; ++m) if (m != i0 && (i1 < 0 || p[m] > p[i1])) i1 = m;
    float w0 = p[i0], w1 = p[i1];
    float inv = 1.f / (w0 + w1 + 1e-8f);
    w0 *= inv; w1 *= inv;
    float v = w0 * cspec[((size_t)i0 * B_ + b) * K_ * SD_ + rem]
            + w1 * cspec[((size_t)i1 * B_ + b) * K_ * SD_ + rem];
    out[(size_t)b * (2 * K_ * SD_) + (size_t)K_ * SD_ + rem] = v;
}

// ---------------- host driver ----------------
extern "C" void kernel_launch(void* const* d_in, const int* in_sizes, int n_in,
                              void* d_out, int out_size, void* d_ws, size_t ws_size,
                              hipStream_t stream)
{
    (void)in_sizes; (void)n_in; (void)out_size; (void)ws_size;
    const float* h = (const float*)d_in[0];
    float* out = (float*)d_out;

    // carve workspace (256B aligned)
    uintptr_t p = (uintptr_t)d_ws;
    auto take = [&](size_t bytes) -> void* {
        void* r = (void*)p; p += (bytes + 255) & ~(size_t)255; return r;
    };
    unsigned short* hB   = (unsigned short*)take((size_t)R_ * TD_ * 2);
    unsigned short* dwB  = (unsigned short*)take((size_t)TD_ * BD_ * 2);   // transposed BD_ x TD_
    unsigned short* uwB  = (unsigned short*)take((size_t)BD_ * SD_ * 2);   // transposed SD_ x BD_
    unsigned short* wkB  = (unsigned short*)take((size_t)SD_ * SD_ * 2);   // transposed
    unsigned short* wvB  = (unsigned short*)take((size_t)SD_ * SD_ * 2);   // transposed
    unsigned short* owB  = (unsigned short*)take((size_t)SD_ * SD_ * 2);   // transposed
    unsigned short* X1   = (unsigned short*)take((size_t)R_ * BD_ * 2);
    float*          X2f  = (float*)take((size_t)R_ * SD_ * 4);
    unsigned short* X2   = (unsigned short*)take((size_t)R_ * SD_ * 2);
    unsigned short* KP   = (unsigned short*)X2f;            // reuse: X2f dead after LN
    unsigned short* VP   = KP + (size_t)R_ * SD_;
    float*          QP   = (float*)take((size_t)K_ * SD_ * 4);
    float*          CTX  = (float*)take((size_t)B_ * K_ * SD_ * 4);
    unsigned short* CTXB = (unsigned short*)take((size_t)B_ * K_ * SD_ * 2);
    float*          CO   = (float*)take((size_t)B_ * K_ * SD_ * 4);
    float*          csp  = (float*)take((size_t)M_ * B_ * K_ * SD_ * 4);
    float*          pooled = (float*)take((size_t)B_ * TD_ * 4);
    float*          probs  = (float*)take((size_t)B_ * M_ * 4);

    cvt_bf16<<<(R_ * TD_) / 256, 256, 0, stream>>>(h, hB, R_ * TD_);

    for (int a = 0; a < 5; ++a) {
        const float *dw,*db,*uw,*ub,*lgm,*lbm,*qm,*wq,*wk,*wv,*ow,*bq,*bk,*bv,*ob,*pg,*pb;
        if (a == 0) {
            dw = (const float*)d_in[1];  db = (const float*)d_in[2];
            uw = (const float*)d_in[3];  ub = (const float*)d_in[4];
            lgm= (const float*)d_in[5];  lbm= (const float*)d_in[6];
            qm = (const float*)d_in[7];
            wq = (const float*)d_in[8];  wk = (const float*)d_in[9];
            wv = (const float*)d_in[10]; ow = (const float*)d_in[11];
            bq = (const float*)d_in[12]; bk = (const float*)d_in[13];
            bv = (const float*)d_in[14]; ob = (const float*)d_in[15];
            pg = (const float*)d_in[16]; pb = (const float*)d_in[17];
        } else {
            size_t m = (size_t)(a - 1);
            dw = (const float*)d_in[18] + m * TD_ * BD_;
            db = (const float*)d_in[19] + m * BD_;
            uw = (const float*)d_in[20] + m * BD_ * SD_;
            ub = (const float*)d_in[21] + m * SD_;
            lgm= (const float*)d_in[22] + m * SD_;
            lbm= (const float*)d_in[23] + m * SD_;
            qm = (const float*)d_in[24] + m * K_ * SD_;
            wq = (const float*)d_in[25] + m * SD_ * SD_;
            wk = (const float*)d_in[26] + m * SD_ * SD_;
            wv = (const float*)d_in[27] + m * SD_ * SD_;
            ow = (const float*)d_in[28] + m * SD_ * SD_;
            bq = (const float*)d_in[29] + m * SD_;
            bk = (const float*)d_in[30] + m * SD_;
            bv = (const float*)d_in[31] + m * SD_;
            ob = (const float*)d_in[32] + m * SD_;
            pg = (const float*)d_in[33] + m * SD_;
            pb = (const float*)d_in[34] + m * SD_;
        }
        // weight conversions (bf16, transposed to N x K for linear DMA staging)
        cvt_bf16_t<<<dim3(BD_ / 32, TD_ / 32), 256, 0, stream>>>(dw, dwB, TD_, BD_);
        cvt_bf16_t<<<dim3(SD_ / 32, BD_ / 32), 256, 0, stream>>>(uw, uwB, BD_, SD_);
        cvt_bf16_t<<<dim3(SD_ / 32, SD_ / 32), 256, 0, stream>>>(wk, wkB, SD_, SD_);
        cvt_bf16_t<<<dim3(SD_ / 32, SD_ / 32), 256, 0, stream>>>(wv, wvB, SD_, SD_);
        cvt_bf16_t<<<dim3(SD_ / 32, SD_ / 32), 256, 0, stream>>>(ow, owB, SD_, SD_);

        // x1 = gelu(h @ dw + db)        (bf16 out)
        wmma_gemm<1,1><<<dim3(BD_ / BN, R_ / BM), 256, 0, stream>>>(
            hB, dwB, db, nullptr, X1, BD_, TD_);
        // x2f = x1 @ uw + ub            (fp32 out, LN next)
        wmma_gemm<0,0><<<dim3(SD_ / BN, R_ / BM), 256, 0, stream>>>(
            X1, uwB, ub, X2f, nullptr, SD_, BD_);
        // x2 = LN(x2f)                  (bf16)
        ln_bf16<<<R_, 256, 0, stream>>>(X2f, lgm, lbm, X2);
        // kp = x2 @ wk + bk ; vp = x2 @ wv + bv   (bf16, reuse X2f storage)
        wmma_gemm<0,1><<<dim3(SD_ / BN, R_ / BM), 256, 0, stream>>>(
            X2, wkB, bk, nullptr, KP, SD_, SD_);
        wmma_gemm<0,1><<<dim3(SD_ / BN, R_ / BM), 256, 0, stream>>>(
            X2, wvB, bv, nullptr, VP, SD_, SD_);
        // qp = q @ wq + bq (tiny, fp32)
        qproj<<<(K_ * SD_) / 256, 256, 0, stream>>>(qm, wq, bq, QP);
        // attention -> CTX fp32
        attn<<<B_ * H_, 256, 0, stream>>>(QP, KP, VP, CTX);
        // out-proj: CO = ctx @ ow + ob  (fp32 out)
        cvt_bf16<<<(B_ * K_ * SD_) / 256, 256, 0, stream>>>(CTX, CTXB, B_ * K_ * SD_);
        wmma_gemm<0,0><<<dim3(SD_ / BN, (B_ * K_) / BM), 256, 0, stream>>>(
            CTXB, owB, ob, CO, nullptr, SD_, SD_);
        // LN(CO + q) -> c_g (into out) or c_spec scratch
        if (a == 0)
            ln_res<<<B_ * K_, 256, 0, stream>>>(CO, qm, pg, pb, out, 2 * K_ * SD_);
        else
            ln_res<<<B_ * K_, 256, 0, stream>>>(CO, qm, pg, pb,
                csp + (size_t)(a - 1) * B_ * K_ * SD_, K_ * SD_);
    }

    // router + mixture
    pool_mean<<<(B_ * TD_) / 256, 256, 0, stream>>>(h, pooled);
    router<<<B_, 512, 0, stream>>>(pooled,
        (const float*)d_in[35], (const float*)d_in[36],
        (const float*)d_in[37], (const float*)d_in[38],
        probs, out + (size_t)B_ * 2 * K_ * SD_);
    combine<<<(B_ * K_ * SD_) / 256, 256, 0, stream>>>(probs, csp, out);
}
